// corss_attention_parallel_block_37460704756390
// MI455X (gfx1250) — compile-verified, matrix operand order checked by reading, not run on hardware
//
#include <hip/hip_runtime.h>
#include <hip/hip_bf16.h>

// ---------------------------------------------------------------------------
// CDNA5 / gfx1250 transformer block (bf16 WMMA everywhere dense math happens)
//   wave32, V_WMMA_F32_16X16X32_BF16, LDS-staged flash attention,
//   TDM tensor_load_to_lds for K tiles + async global->LDS for V tiles
//   (both feature-guarded; plain path otherwise)
// ---------------------------------------------------------------------------

typedef __bf16 bf16_t;
typedef bf16_t v16bf __attribute__((ext_vector_type(16)));
typedef float  v8f   __attribute__((ext_vector_type(8)));

#define DEVINL __device__ __forceinline__

DEVINL unsigned short f2bf(float f) {
  unsigned u = __float_as_uint(f);
  unsigned r = u + 0x7FFFu + ((u >> 16) & 1u);   // round-to-nearest-even
  return (unsigned short)(r >> 16);
}

// ----------------------------- CDNA5 data movers ---------------------------

#define AS_GLOBAL __attribute__((address_space(1)))
#define AS_LDS    __attribute__((address_space(3)))

DEVINL unsigned lds_byte_offset(const void* p) {
  return (unsigned)(unsigned long long)(AS_LDS const void*)p;
}

#if __has_builtin(__builtin_amdgcn_tensor_load_to_lds) && \
    __has_builtin(__builtin_amdgcn_s_wait_tensorcnt)
#define HAVE_TDM 1
typedef unsigned int u32x4 __attribute__((ext_vector_type(4)));
typedef int          i32x4 __attribute__((ext_vector_type(4)));
typedef int          i32x8 __attribute__((ext_vector_type(8)));

// 2D tile load: rows x cols elements of 2 bytes, row stride in elements.
// D# packed per CDNA5 ISA 8.3/8.4 (group0: count/lds/global addr/type=2,
// group1: data_size=2B, tensor dims == tile dims, dim0 stride).
DEVINL void tdm_load_tile_2d(unsigned lds_off, unsigned long long gaddr,
                             int stride_elems, int rows, int cols) {
  u32x4 g0;
  g0[0] = 1u;                                   // count=1, user descriptor
  g0[1] = lds_off;                              // lds_addr (bytes)
  g0[2] = (unsigned)(gaddr & 0xFFFFFFFFull);    // global_addr[31:0]
  g0[3] = (unsigned)((gaddr >> 32) & 0x1FFFFFFull) | (2u << 30);  // [56:32]|type=2
  i32x8 g1;
  g1[0] = 1 << 16;                              // data_size=1 -> 2-byte elems
  g1[1] = (cols & 0xFFFF) << 16;                // tensor_dim0[15:0] @ bits63:48
  g1[2] = ((cols >> 16) & 0xFFFF) | ((rows & 0xFFFF) << 16);  // dim0 hi | dim1 lo
  g1[3] = ((rows >> 16) & 0xFFFF) | ((cols & 0xFFFF) << 16);  // dim1 hi | tile_dim0
  g1[4] = rows & 0xFFFF;                        // tile_dim1 (tile_dim2=0)
  g1[5] = stride_elems;                         // tensor_dim0_stride[31:0]
  g1[6] = 0;                                    // stride hi | dim1_stride lo
  g1[7] = 0;
  i32x4 z4 = {0, 0, 0, 0};
#if defined(__clang_major__) && __clang_major__ >= 23
  i32x8 z8 = {0, 0, 0, 0, 0, 0, 0, 0};
  __builtin_amdgcn_tensor_load_to_lds(g0, g1, z4, z4, z8, 0);
#else
  __builtin_amdgcn_tensor_load_to_lds(g0, g1, z4, z4, 0);
#endif
}
#endif  // HAVE_TDM

#if __has_builtin(__builtin_amdgcn_global_load_async_to_lds_b128)
#define HAVE_ASYNC 1
typedef int v4i_vs __attribute__((vector_size(4 * sizeof(int))));
DEVINL void async_ld_b128(const void* gsrc, void* ldst) {
  __builtin_amdgcn_global_load_async_to_lds_b128(
      (AS_GLOBAL v4i_vs*)gsrc, (AS_LDS v4i_vs*)ldst, 0, 0);
}
DEVINL void wait_async0() {
#if __has_builtin(__builtin_amdgcn_s_wait_asynccnt)
  __builtin_amdgcn_s_wait_asynccnt(0);
#else
  asm volatile("s_wait_asynccnt 0x0" ::: "memory");
#endif
}
#endif  // HAVE_ASYNC

// ============================ pack / convert kernels ========================

__global__ void k_f32_to_bf16(const float* __restrict__ s,
                              unsigned short* __restrict__ d, int n) {
  int i = blockIdx.x * blockDim.x + threadIdx.x;
  if (i < n) d[i] = f2bf(s[i]);
}

// W (K x N) f32 -> Wt (N x K) bf16  (so WMMA B-fragment lanes load contiguous K)
__global__ void k_transpose_bf16(const float* __restrict__ s,
                                 unsigned short* __restrict__ d, int K, int N) {
  int i = blockIdx.x * blockDim.x + threadIdx.x;
  if (i >= N * K) return;
  int n = i / K, k = i - n * K;
  d[i] = f2bf(s[(size_t)k * N + n]);
}

// ca_Wk/Wv (512 x 1024) -> folded over the 16 group blocks -> (64 x 512) bf16
// (reference einsum sums over the group axis, so K_eff = ctx @ sum_g Wk[:,g,:])
__global__ void k_fold_group(const float* __restrict__ s,
                             unsigned short* __restrict__ d) {
  int i = blockIdx.x * blockDim.x + threadIdx.x;
  if (i >= 64 * 512) return;
  int dd = i >> 9, k = i & 511;
  float acc = 0.f;
#pragma unroll
  for (int g = 0; g < 16; ++g) acc += s[(size_t)k * 1024 + g * 64 + dd];
  d[i] = f2bf(acc);
}

__global__ void k_fold_bias(const float* __restrict__ s, float* __restrict__ d) {
  int i = threadIdx.x;  // 64 threads
  float acc = 0.f;
#pragma unroll
  for (int g = 0; g < 16; ++g) acc += s[g * 64 + i];
  d[i] = acc;
}

// sa_W{q,k,v} (16,1024,64) -> column-concatenated transposed (1024 x 1024) bf16
__global__ void k_pack_sa_qkv(const float* __restrict__ s,
                              unsigned short* __restrict__ d) {
  int i = blockIdx.x * blockDim.x + threadIdx.x;   // over 1024*1024
  int n = i >> 10, m = i & 1023;
  int h = n >> 6, dd = n & 63;
  d[i] = f2bf(s[((size_t)h << 16) + (m << 6) + dd]);
}

// sa_Wh (16,64,64) -> block-diagonal transposed (1024 x 1024) bf16
__global__ void k_pack_blockdiag(const float* __restrict__ s,
                                 unsigned short* __restrict__ d) {
  int i = blockIdx.x * blockDim.x + threadIdx.x;
  int n = i >> 10, k = i & 1023;
  int h = n >> 6;
  float v = 0.f;
  if ((k >> 6) == h) v = s[((size_t)h << 12) + ((k & 63) << 6) + (n & 63)];
  d[i] = f2bf(v);
}

// ============================ layernorm (f32 in, bf16 out) ==================

__global__ __launch_bounds__(256) void ln_bf16(const float* __restrict__ x,
                                               const float* __restrict__ g,
                                               const float* __restrict__ bta,
                                               unsigned short* __restrict__ y) {
  __shared__ float sh[512];
  const int row = blockIdx.x;
  const float* xr = x + (size_t)row * 1024;
  float s = 0.f, s2 = 0.f;
  for (int i = threadIdx.x; i < 1024; i += 256) {
    float v = xr[i]; s += v; s2 += v * v;
  }
  sh[threadIdx.x] = s; sh[256 + threadIdx.x] = s2;
  __syncthreads();
  for (int st = 128; st > 0; st >>= 1) {
    if (threadIdx.x < st) {
      sh[threadIdx.x]       += sh[threadIdx.x + st];
      sh[256 + threadIdx.x] += sh[256 + threadIdx.x + st];
    }
    __syncthreads();
  }
  const float mean = sh[0]   * (1.f / 1024.f);
  const float var  = sh[256] * (1.f / 1024.f) - mean * mean;
  const float rstd = rsqrtf(var + 1e-5f);
  unsigned short* yr = y + (size_t)row * 1024;
  for (int i = threadIdx.x; i < 1024; i += 256)
    yr[i] = f2bf((xr[i] - mean) * rstd * g[i] + bta[i]);
}

__global__ void k_add2(const float* __restrict__ a, const float* __restrict__ b,
                       float* __restrict__ c, int n) {
  int i = blockIdx.x * blockDim.x + threadIdx.x;
  if (i < n) c[i] = a[i] + b[i];
}

// ============================ bf16 WMMA GEMM ================================
// Y(MxN) = act( X(MxK) @ W(KxN) + bias ),  W pre-transposed as Bt (N x K).
// Block: BWM x BWN waves; each wave computes a (WM*16) x (WN*16) tile with
// v_wmma_f32_16x16x32_bf16.  EPI: 0=none, 1=SiLU.  OUTF32: 0=bf16, 1=f32.

template <int BWM, int BWN, int WM, int WN, int EPI, int OUTF32>
__global__ __launch_bounds__(32 * BWM * BWN) void gemm_wmma(
    const unsigned short* __restrict__ A, const unsigned short* __restrict__ Bt,
    const float* __restrict__ bias, void* __restrict__ Cout,
    int M, int N, int K) {
  const int tid  = threadIdx.x;
  const int lane = tid & 31, wid = tid >> 5;
  const int wm = wid / BWN, wn = wid % BWN;
  const int lm = lane & 15, lh = lane >> 4;
  const int m0 = blockIdx.y * (BWM * WM * 16) + wm * (WM * 16);
  const int n0 = blockIdx.x * (BWN * WN * 16) + wn * (WN * 16);

  const v8f zero = {0, 0, 0, 0, 0, 0, 0, 0};
  v8f acc[WM][WN];
#pragma unroll
  for (int i = 0; i < WM; ++i)
#pragma unroll
    for (int j = 0; j < WN; ++j) acc[i][j] = zero;

  for (int k0 = 0; k0 < K; k0 += 32) {
    v16bf af[WM], bfr[WN];
#pragma unroll
    for (int i = 0; i < WM; ++i)
      af[i] = *(const v16bf*)(A + (size_t)(m0 + i * 16 + lm) * K + k0 + lh * 16);
#pragma unroll
    for (int j = 0; j < WN; ++j)
      bfr[j] = *(const v16bf*)(Bt + (size_t)(n0 + j * 16 + lm) * K + k0 + lh * 16);
#pragma unroll
    for (int i = 0; i < WM; ++i)
#pragma unroll
      for (int j = 0; j < WN; ++j)
        acc[i][j] = __builtin_amdgcn_wmma_f32_16x16x32_bf16(
            false, af[i], false, bfr[j], (short)0, acc[i][j], false, false);
  }

#pragma unroll
  for (int i = 0; i < WM; ++i) {
#pragma unroll
    for (int j = 0; j < WN; ++j) {
      const int col = n0 + j * 16 + lm;
      const float bv = bias[col];
#pragma unroll
      for (int r = 0; r < 8; ++r) {
        const int row = m0 + i * 16 + r + (lh << 3);
        float v = acc[i][j][r] + bv;
        if constexpr (EPI == 1) v = v / (1.f + __expf(-v));   // SiLU
        if constexpr (OUTF32 != 0)
          ((float*)Cout)[(size_t)row * N + col] = v;
        else
          ((unsigned short*)Cout)[(size_t)row * N + col] = f2bf(v);
      }
    }
  }
}

// ============================ flash attention (WMMA) ========================
// One block = 8 waves x 16 queries = 128 queries of one (b,h).  K/V tiles of
// 64 keys staged in LDS:
//   * K tile: TDM tensor_load_to_lds (wave 0 issues; s_wait_tensorcnt) when
//     available, else global->VGPR->LDS copy.
//   * V tile: async global->LDS rows (ASYNCcnt) then per-wave LDS transpose
//     when available, else global->VGPR scatter.
// Online softmax on C-fragment row stats; P re-fragmented via per-wave LDS.

__global__ __launch_bounds__(256) void attn_wmma(
    const unsigned short* __restrict__ Q, const unsigned short* __restrict__ K,
    const unsigned short* __restrict__ V, unsigned short* __restrict__ O,
    float scale, int T, int Vlen, int kv_bstride, int kv_vstride,
    int kv_hstride) {
  __shared__ __align__(32) unsigned short Kt[64 * 64];       // key x d
  __shared__ __align__(32) unsigned short Vt[64 * 64];       // d x key
  __shared__ __align__(32) unsigned short Pt[8 * 16 * 64];   // per-wave P strip
#if defined(HAVE_ASYNC)
  __shared__ __align__(32) unsigned short Vrow[64 * 64];     // key x d staging
#endif

  const int tid = threadIdx.x;
  const int lane = tid & 31, w = tid >> 5;
  const int lm = lane & 15, lh = lane >> 4;
  const int b = blockIdx.y >> 4, h = blockIdx.y & 15;
  const int q0 = blockIdx.x * 128 + w * 16;

  const unsigned short* Kbp = K + (size_t)b * kv_bstride + (size_t)h * kv_hstride;
  const unsigned short* Vbp = V + (size_t)b * kv_bstride + (size_t)h * kv_hstride;

  const v8f zero = {0, 0, 0, 0, 0, 0, 0, 0};
  v8f oacc[4] = {zero, zero, zero, zero};
  float m_[8], l_[8];
#pragma unroll
  for (int r = 0; r < 8; ++r) { m_[r] = -1e30f; l_[r] = 0.f; }

  const int ldk = tid >> 2;           // key row this thread stages
  const int ldd = (tid & 3) * 16;     // d offset (16 elements = 32B)

  for (int v0 = 0; v0 < Vlen; v0 += 64) {
    // ---- stage K tile (key x d, row-major) ----
#if defined(HAVE_TDM)
    if (tid < 32) {
      tdm_load_tile_2d(lds_byte_offset(&Kt[0]),
                       (unsigned long long)(const void*)
                           (Kbp + (size_t)v0 * kv_vstride),
                       kv_vstride, 64, 64);
      __builtin_amdgcn_s_wait_tensorcnt(0);
    }
#else
    {
      const unsigned short* ks = Kbp + (size_t)(v0 + ldk) * kv_vstride + ldd;
      const uint4* k4 = (const uint4*)ks;
      *(uint4*)&Kt[ldk * 64 + ldd]     = k4[0];
      *(uint4*)&Kt[ldk * 64 + ldd + 8] = k4[1];
    }
#endif
    // ---- stage V tile transposed (d x key) ----
#if defined(HAVE_ASYNC)
    {
      const unsigned short* vs = Vbp + (size_t)(v0 + ldk) * kv_vstride + ldd;
      unsigned short* vd = &Vrow[ldk * 64 + ldd];
      async_ld_b128(vs, vd);
      async_ld_b128(vs + 8, vd + 8);
      wait_async0();                         // own wave's rows are in LDS
#pragma unroll
      for (int dd = 0; dd < 16; ++dd)        // transpose only own chunk
        Vt[(ldd + dd) * 64 + ldk] = vd[dd];
    }
#else
    {
      const unsigned short* vs = Vbp + (size_t)(v0 + ldk) * kv_vstride + ldd;
#pragma unroll
      for (int dd = 0; dd < 16; ++dd) Vt[(ldd + dd) * 64 + ldk] = vs[dd];
    }
#endif
    __syncthreads();

    // S = Q K^T  (16 queries x 64 keys) via 2 K-steps x 4 key tiles
    v8f s[4] = {zero, zero, zero, zero};
#pragma unroll
    for (int kd = 0; kd < 64; kd += 32) {
      v16bf aq = *(const v16bf*)(Q + ((size_t)(b * T + q0 + lm)) * 1024 +
                                 h * 64 + kd + lh * 16);
#pragma unroll
      for (int j = 0; j < 4; ++j) {
        v16bf bk = *(const v16bf*)&Kt[(j * 16 + lm) * 64 + kd + lh * 16];
        s[j] = __builtin_amdgcn_wmma_f32_16x16x32_bf16(
            false, aq, false, bk, (short)0, s[j], false, false);
      }
    }

    // online softmax per row (C-frag: VGPR r = row r / r+8 per lane half)
#pragma unroll
    for (int r = 0; r < 8; ++r) {
      float mx = -1e30f;
#pragma unroll
      for (int j = 0; j < 4; ++j) { s[j][r] *= scale; mx = fmaxf(mx, s[j][r]); }
#pragma unroll
      for (int d = 1; d < 16; d <<= 1) mx = fmaxf(mx, __shfl_xor(mx, d, 32));
      float mn = fmaxf(m_[r], mx);
      float corr = __expf(m_[r] - mn);
      m_[r] = mn; l_[r] *= corr;
#pragma unroll
      for (int j = 0; j < 4; ++j) oacc[j][r] *= corr;
      float rs = 0.f;
      const int prow = (w * 16 + r + (lh << 3)) * 64;
#pragma unroll
      for (int j = 0; j < 4; ++j) {
        float p = __expf(s[j][r] - mn);
        rs += p;
        Pt[prow + j * 16 + lm] = f2bf(p);
      }
#pragma unroll
      for (int d = 1; d < 16; d <<= 1) rs += __shfl_xor(rs, d, 32);
      l_[r] += rs;
    }

    // O += P V  (P from per-wave LDS strip as A-fragments, V^T as B)
#pragma unroll
    for (int kv = 0; kv < 64; kv += 32) {
      v16bf pa = *(const v16bf*)&Pt[(w * 16 + lm) * 64 + kv + lh * 16];
#pragma unroll
      for (int j = 0; j < 4; ++j) {
        v16bf vb = *(const v16bf*)&Vt[(j * 16 + lm) * 64 + kv + lh * 16];
        oacc[j] = __builtin_amdgcn_wmma_f32_16x16x32_bf16(
            false, pa, false, vb, (short)0, oacc[j], false, false);
      }
    }
    __syncthreads();
  }

#pragma unroll
  for (int r = 0; r < 8; ++r) {
    const float inv = 1.f / l_[r];
    const int row = q0 + r + (lh << 3);
    const size_t base = ((size_t)(b * T + row)) * 1024 + h * 64;
#pragma unroll
    for (int j = 0; j < 4; ++j)
      O[base + j * 16 + lm] = f2bf(oacc[j][r] * inv);
  }
}

// ============================ host-side orchestration =======================

extern "C" void kernel_launch(void* const* d_in, const int* in_sizes, int n_in,
                              void* d_out, int out_size, void* d_ws,
                              size_t ws_size, hipStream_t stream) {
  (void)in_sizes; (void)n_in; (void)out_size; (void)ws_size;
  const float* x      = (const float*)d_in[0];
  const float* cond   = (const float*)d_in[1];
  const float* adj    = (const float*)d_in[2];
  const float* ca_Wq  = (const float*)d_in[3];
  const float* ca_bq  = (const float*)d_in[4];
  const float* ca_Wk  = (const float*)d_in[5];
  const float* ca_bk  = (const float*)d_in[6];
  const float* ca_Wv  = (const float*)d_in[7];
  const float* ca_bv  = (const float*)d_in[8];
  const float* ca_Wo  = (const float*)d_in[9];
  const float* ca_bo  = (const float*)d_in[10];
  const float* aa_Wq  = (const float*)d_in[11];
  const float* aa_bq  = (const float*)d_in[12];
  const float* aa_Wk  = (const float*)d_in[13];
  const float* aa_bk  = (const float*)d_in[14];
  const float* aa_Wv  = (const float*)d_in[15];
  const float* aa_bv  = (const float*)d_in[16];
  const float* aa_Wo  = (const float*)d_in[17];
  const float* aa_bo  = (const float*)d_in[18];
  const float* sa_Wq  = (const float*)d_in[19];
  const float* sa_bq  = (const float*)d_in[20];
  const float* sa_Wk  = (const float*)d_in[21];
  const float* sa_bk  = (const float*)d_in[22];
  const float* sa_Wv  = (const float*)d_in[23];
  const float* sa_bv  = (const float*)d_in[24];
  const float* sa_Wh  = (const float*)d_in[25];
  const float* sa_bh  = (const float*)d_in[26];
  const float* sa_Wp  = (const float*)d_in[27];
  const float* sa_bp  = (const float*)d_in[28];
  const float* ff_W1  = (const float*)d_in[29];
  const float* ff_b1  = (const float*)d_in[30];
  const float* ff_W2  = (const float*)d_in[31];
  const float* ff_b2  = (const float*)d_in[32];
  const float* ln11_g = (const float*)d_in[33];
  const float* ln11_b = (const float*)d_in[34];
  const float* ln12_g = (const float*)d_in[35];
  const float* ln12_b = (const float*)d_in[36];
  const float* ln2_g  = (const float*)d_in[37];
  const float* ln2_b  = (const float*)d_in[38];
  const float* ln3_g  = (const float*)d_in[39];
  const float* ln3_b  = (const float*)d_in[40];

  char* ws = (char*)d_ws;
  const size_t MB = 1024ull * 1024ull;
  unsigned short* hA    = (unsigned short*)(ws + 0 * MB);     // 8 MB
  unsigned short* Qb    = (unsigned short*)(ws + 8 * MB);     // 8 MB
  unsigned short* Kb    = (unsigned short*)(ws + 16 * MB);    // 8 MB
  unsigned short* Vb    = (unsigned short*)(ws + 24 * MB);    // 8 MB
  unsigned short* Ob    = (unsigned short*)(ws + 32 * MB);    // 8 MB
  unsigned short* O2b   = (unsigned short*)(ws + 40 * MB);    // 8 MB
  float*          Pf    = (float*)(ws + 48 * MB);             // 16 MB
  float*          xacc  = (float*)(ws + 64 * MB);             // 16 MB
  unsigned short* condb = (unsigned short*)(ws + 80 * MB);    // 4 MB
  unsigned short* adjb  = (unsigned short*)(ws + 84 * MB);    // 4 MB
  unsigned short* wCAQ  = (unsigned short*)(ws + 88 * MB);
  unsigned short* wCAO  = (unsigned short*)(ws + 90 * MB);
  unsigned short* wAAQ  = (unsigned short*)(ws + 92 * MB);
  unsigned short* wAAO  = (unsigned short*)(ws + 94 * MB);
  unsigned short* wSAQ  = (unsigned short*)(ws + 96 * MB);
  unsigned short* wSAK  = (unsigned short*)(ws + 98 * MB);
  unsigned short* wSAV  = (unsigned short*)(ws + 100 * MB);
  unsigned short* wSAH  = (unsigned short*)(ws + 102 * MB);
  unsigned short* wSAP  = (unsigned short*)(ws + 104 * MB);
  unsigned short* wFF1  = (unsigned short*)(ws + 106 * MB);   // 4 MB
  unsigned short* wFF2  = (unsigned short*)(ws + 110 * MB);   // 4 MB
  unsigned short* wCAK  = (unsigned short*)(ws + 114 * MB);
  unsigned short* wCAV  = (unsigned short*)(ws + 114 * MB + 64 * 1024);
  unsigned short* wAAK  = (unsigned short*)(ws + 114 * MB + 128 * 1024);
  unsigned short* wAAV  = (unsigned short*)(ws + 114 * MB + 192 * 1024);
  float*          bCAK  = (float*)(ws + 114 * MB + 256 * 1024);
  float*          bCAV  = (float*)(ws + 114 * MB + 256 * 1024 + 256);
  unsigned short* Gb    = Qb;  // FF hidden (16 MB) aliases then-dead Qb+Kb

  const int TPB = 256;
  const int BT = 4096, NHID = 1024, NE = 512, DH = 64;
  const int nTok = BT * NHID;

  // ---- weight / activation packing (f32 -> bf16, transposes, foldings) ----
  k_f32_to_bf16<<<(BT * NE + 255) / 256, TPB, 0, stream>>>(cond, condb, BT * NE);
  k_f32_to_bf16<<<(BT * NE + 255) / 256, TPB, 0, stream>>>(adj, adjb, BT * NE);
  k_transpose_bf16<<<4096, TPB, 0, stream>>>(ca_Wq, wCAQ, 1024, 1024);
  k_transpose_bf16<<<4096, TPB, 0, stream>>>(ca_Wo, wCAO, 1024, 1024);
  k_transpose_bf16<<<4096, TPB, 0, stream>>>(aa_Wq, wAAQ, 1024, 1024);
  k_transpose_bf16<<<4096, TPB, 0, stream>>>(aa_Wo, wAAO, 1024, 1024);
  k_transpose_bf16<<<4096, TPB, 0, stream>>>(sa_Wp, wSAP, 1024, 1024);
  k_transpose_bf16<<<8192, TPB, 0, stream>>>(ff_W1, wFF1, 1024, 2048);
  k_transpose_bf16<<<8192, TPB, 0, stream>>>(ff_W2, wFF2, 2048, 1024);
  k_transpose_bf16<<<128, TPB, 0, stream>>>(aa_Wk, wAAK, 512, 64);
  k_transpose_bf16<<<128, TPB, 0, stream>>>(aa_Wv, wAAV, 512, 64);
  k_fold_group<<<128, TPB, 0, stream>>>(ca_Wk, wCAK);
  k_fold_group<<<128, TPB, 0, stream>>>(ca_Wv, wCAV);
  k_fold_bias<<<1, 64, 0, stream>>>(ca_bk, bCAK);
  k_fold_bias<<<1, 64, 0, stream>>>(ca_bv, bCAV);
  k_pack_sa_qkv<<<4096, TPB, 0, stream>>>(sa_Wq, wSAQ);
  k_pack_sa_qkv<<<4096, TPB, 0, stream>>>(sa_Wk, wSAK);
  k_pack_sa_qkv<<<4096, TPB, 0, stream>>>(sa_Wv, wSAV);
  k_pack_blockdiag<<<4096, TPB, 0, stream>>>(sa_Wh, wSAH);

  dim3 gBig(4, 32);   // N=1024 (4 x 256-col blocks), M=4096 (32 x 128-row)
  dim3 gFF1(8, 32);   // N=2048
  dim3 gNar(1, 16);   // N=64, M=4096 (16 x 256-row)
  dim3 gAtt(8, 64);   // T/128 query tiles, B*NH heads
  const int nAdd = (nTok + 255) / 256;

  // ===== cross-attention on cond (group-summed K/V -> MQA, scale 1/8) =====
  ln_bf16<<<4096, TPB, 0, stream>>>(x, ln11_g, ln11_b, hA);
  gemm_wmma<2, 4, 4, 4, 0, 0><<<gBig, TPB, 0, stream>>>(hA, wCAQ, ca_bq, Qb, BT, NHID, NHID);
  gemm_wmma<8, 1, 2, 4, 0, 0><<<gNar, TPB, 0, stream>>>(condb, wCAK, bCAK, Kb, BT, DH, NE);
  gemm_wmma<8, 1, 2, 4, 0, 0><<<gNar, TPB, 0, stream>>>(condb, wCAV, bCAV, Vb, BT, DH, NE);
  attn_wmma<<<gAtt, TPB, 0, stream>>>(Qb, Kb, Vb, Ob, 0.125f, 1024, 1024, 1024 * 64, 64, 0);
  gemm_wmma<2, 4, 4, 4, 0, 1><<<gBig, TPB, 0, stream>>>(Ob, wCAO, ca_bo, Pf, BT, NHID, NHID);
  k_add2<<<nAdd, TPB, 0, stream>>>(x, Pf, xacc, nTok);

  // ===== cross-attention on adj (MQA, scale 1/8) =====
  ln_bf16<<<4096, TPB, 0, stream>>>(x, ln12_g, ln12_b, hA);
  gemm_wmma<2, 4, 4, 4, 0, 0><<<gBig, TPB, 0, stream>>>(hA, wAAQ, aa_bq, Qb, BT, NHID, NHID);
  gemm_wmma<8, 1, 2, 4, 0, 0><<<gNar, TPB, 0, stream>>>(adjb, wAAK, aa_bk, Kb, BT, DH, NE);
  gemm_wmma<8, 1, 2, 4, 0, 0><<<gNar, TPB, 0, stream>>>(adjb, wAAV, aa_bv, Vb, BT, DH, NE);
  attn_wmma<<<gAtt, TPB, 0, stream>>>(Qb, Kb, Vb, Ob, 0.125f, 1024, 1024, 1024 * 64, 64, 0);
  gemm_wmma<2, 4, 4, 4, 0, 1><<<gBig, TPB, 0, stream>>>(Ob, wAAO, aa_bo, Pf, BT, NHID, NHID);
  k_add2<<<nAdd, TPB, 0, stream>>>(xacc, Pf, xacc, nTok);

  // ===== per-head self-attention (scale 1/sqrt(NHID) = 1/32) =====
  ln_bf16<<<4096, TPB, 0, stream>>>(xacc, ln2_g, ln2_b, hA);
  gemm_wmma<2, 4, 4, 4, 0, 0><<<gBig, TPB, 0, stream>>>(hA, wSAQ, sa_bq, Qb, BT, NHID, NHID);
  gemm_wmma<2, 4, 4, 4, 0, 0><<<gBig, TPB, 0, stream>>>(hA, wSAK, sa_bk, Kb, BT, NHID, NHID);
  gemm_wmma<2, 4, 4, 4, 0, 0><<<gBig, TPB, 0, stream>>>(hA, wSAV, sa_bv, Vb, BT, NHID, NHID);
  attn_wmma<<<gAtt, TPB, 0, stream>>>(Qb, Kb, Vb, Ob, 0.03125f, 1024, 1024, 1024 * 1024, 1024, 64);
  gemm_wmma<2, 4, 4, 4, 0, 0><<<gBig, TPB, 0, stream>>>(Ob, wSAH, sa_bh, O2b, BT, NHID, NHID);
  gemm_wmma<2, 4, 4, 4, 0, 1><<<gBig, TPB, 0, stream>>>(O2b, wSAP, sa_bp, Pf, BT, NHID, NHID);
  k_add2<<<nAdd, TPB, 0, stream>>>(xacc, Pf, xacc, nTok);

  // ===== feed-forward (SiLU) =====
  ln_bf16<<<4096, TPB, 0, stream>>>(xacc, ln3_g, ln3_b, hA);
  gemm_wmma<2, 4, 4, 4, 1, 0><<<gFF1, TPB, 0, stream>>>(hA, wFF1, ff_b1, Gb, BT, 2048, 1024);
  gemm_wmma<2, 4, 4, 4, 0, 1><<<gBig, TPB, 0, stream>>>(Gb, wFF2, ff_b2, Pf, BT, NHID, 2048);
  k_add2<<<nAdd, TPB, 0, stream>>>(xacc, Pf, (float*)d_out, nTok);
}
